// LinearReluAttention_43456479101777
// MI455X (gfx1250) — matrix operand order repaired and checked by reading, not compiled, not run on hardware
//
#include <hip/hip_runtime.h>
#include <cstdint>
#include <cstddef>

// ---------------------------------------------------------------------------
// CDNA5 (gfx1250) wave32 WMMA types
// ---------------------------------------------------------------------------
typedef __attribute__((ext_vector_type(16))) _Float16 v16h;
typedef __attribute__((ext_vector_type(8)))  _Float16 v8h;
typedef __attribute__((ext_vector_type(8)))  float    v8f;
typedef __attribute__((ext_vector_type(4)))  int      v4i;

union HFrag { v16h v; v8h h[2]; _Float16 e[16]; };

// ---- async global->LDS staging (gfx1250), with sync fallback --------------
#if defined(__has_builtin)
#if __has_builtin(__builtin_amdgcn_global_load_async_to_lds_b128) && \
    __has_builtin(__builtin_amdgcn_s_wait_asynccnt)
#define HAVE_ASYNC_LDS 1
#endif
#endif
#ifndef HAVE_ASYNC_LDS
#define HAVE_ASYNC_LDS 0
#endif

__device__ __forceinline__ void g2l_b128(const _Float16* g, _Float16* l) {
#if HAVE_ASYNC_LDS
  // param 1 is v4i* per clang diagnostic; order follows global_load_lds:
  // (global source, LDS destination, offset, cpol)
  __builtin_amdgcn_global_load_async_to_lds_b128((v4i*)g, (v4i*)l, 0, 0);
#else
  *(v8h*)l = *(const v8h*)g;
#endif
}

__device__ __forceinline__ void wait_async_le4() {
#if HAVE_ASYNC_LDS
  __builtin_amdgcn_s_wait_asynccnt(4);
#endif
}
__device__ __forceinline__ void wait_async_0() {
#if HAVE_ASYNC_LDS
  __builtin_amdgcn_s_wait_asynccnt(0);
#endif
}

__device__ __forceinline__ v8f wmma_f16(v16h a, v16h b, v8f c) {
  // D = A(16x32 f16) * B(32x16 f16) + C(16x16 f32)
  return __builtin_amdgcn_wmma_f32_16x16x32_f16(false, a, false, b, (short)0, c,
                                                false, false);
}

// A fragment (16x32 f16) from row-major [rows][ld] f16 LDS storage.
// lane<16 -> M=lane, K=k0+[0..15];
// lane>=16 -> M=lane-16, K=k0+[16..31].
__device__ __forceinline__ v16h frag_A(const _Float16* base, int ld, int row0,
                                       int k0, int lane) {
  const int l16 = lane & 15;
  const int hi  = (lane >> 4) & 1;
  const _Float16* p = base + (row0 + l16) * ld + k0 + hi * 16;
  HFrag f;
  f.h[0] = *(const v8h*)(p);
  f.h[1] = *(const v8h*)(p + 8);
  return f.v;
}

// B fragment (32x16 f16, KxN) from row-major [N][ld] storage (row = output
// column N with contiguous K). lane<16: N=lane, K=k0+[0..15]; lane>=16:
// N=lane-16, K=k0+[16..31].
__device__ __forceinline__ v16h frag_B(const _Float16* base, int ld, int n0,
                                       int k0, int lane) {
  const int l16 = lane & 15;
  const int hi  = (lane >> 4) & 1;
  const _Float16* p = base + (n0 + l16) * ld + k0 + hi * 16;
  HFrag f;
  f.h[0] = *(const v8h*)(p);
  f.h[1] = *(const v8h*)(p + 8);
  return f.v;
}

// B fragment sourced from row-major f32 (global), converted in flight.
__device__ __forceinline__ v16h frag_B_f32(const float* base, int ld, int n0,
                                           int k0, int lane) {
  const int l16 = lane & 15;
  const int hi  = (lane >> 4) & 1;
  const float* p = base + (size_t)(n0 + l16) * ld + k0 + hi * 16;
  HFrag f;
#pragma unroll
  for (int g = 0; g < 4; ++g) {
    const float4 s = *(const float4*)(p + g * 4);
    f.e[g * 4 + 0] = (_Float16)s.x;
    f.e[g * 4 + 1] = (_Float16)s.y;
    f.e[g * 4 + 2] = (_Float16)s.z;
    f.e[g * 4 + 3] = (_Float16)s.w;
  }
  return f.v;
}

// ---------------------------------------------------------------------------
// Kernel 0: bulk f32 -> f16 conversion (x, qkv_w, proj_w)
// ---------------------------------------------------------------------------
__global__ __launch_bounds__(256) void f32_to_f16(const float* __restrict__ src,
                                                  _Float16* __restrict__ dst,
                                                  int n4) {
  const int i = blockIdx.x * 256 + threadIdx.x;
  if (i < n4) {
    const float4 s = ((const float4*)src)[i];
    union { _Float16 h[4]; uint2 u; } r;
    r.h[0] = (_Float16)s.x;
    r.h[1] = (_Float16)s.y;
    r.h[2] = (_Float16)s.z;
    r.h[3] = (_Float16)s.w;
    ((uint2*)dst)[i] = r.u;
  }
}

// ---------------------------------------------------------------------------
// Kernels 1 & 5: C[M][N] = A[M][K] @ B[N][K]^T, all-f16 operands in global.
// 256 threads = 8 waves; block tile 128x128, K-step 32; wave tile 32x64
// (2x4 WMMA accumulators -> 8 v_wmma per wave per K-step).
// Double-buffered LDS staging via async global->LDS copies (ASYNCcnt).
// ---------------------------------------------------------------------------
template <typename OutT>
__global__ __launch_bounds__(256) void gemm_xwt_f16(
    const _Float16* __restrict__ A, const _Float16* __restrict__ B,
    OutT* __restrict__ C, int M, int N, int K) {
  __shared__ _Float16 As[2][128][40];  // 128x32 tile + 8-half pad per row
  __shared__ _Float16 Bs[2][128][40];

  const int tid  = threadIdx.x;
  const int lane = tid & 31;
  const int wave = tid >> 5;
  const int l16  = lane & 15;
  const int hi   = (lane >> 4) & 1;
  const int mBlk = blockIdx.y * 128;
  const int nBlk = blockIdx.x * 128;
  const int m0w  = (wave >> 1) * 32;  // 4 waves along M
  const int n0w  = (wave & 1) * 64;   // 2 waves along N

  v8f acc[2][4] = {};
  const int KT = K >> 5;

  // Stage one 128x32 tile of A and B into buffer `buf` (4 async copies/thread).
  auto stage = [&](int buf, int kt) {
    const int k0 = kt << 5;
#pragma unroll
    for (int c = tid; c < 1024; c += 256) {  // 2 x (128 rows x 4 b128-chunks)
      const int isB = c >> 9;
      const int r   = (c & 511) >> 2;
      const int g   = (c & 3) * 8;
      const _Float16* gp =
          (isB ? B + (size_t)(nBlk + r) * K : A + (size_t)(mBlk + r) * K) +
          k0 + g;
      _Float16* lp = isB ? &Bs[buf][r][g] : &As[buf][r][g];
      g2l_b128(gp, lp);
    }
  };

  auto compute = [&](int buf) {
    const v16h a0 = frag_A(&As[buf][0][0], 40, m0w, 0, lane);
    const v16h a1 = frag_A(&As[buf][0][0], 40, m0w + 16, 0, lane);
    v16h b[4];
#pragma unroll
    for (int ni = 0; ni < 4; ++ni)
      b[ni] = frag_B(&Bs[buf][0][0], 40, n0w + 16 * ni, 0, lane);
#pragma unroll
    for (int ni = 0; ni < 4; ++ni) {
      acc[0][ni] = wmma_f16(a0, b[ni], acc[0][ni]);
      acc[1][ni] = wmma_f16(a1, b[ni], acc[1][ni]);
    }
  };

  stage(0, 0);
  for (int kt = 0; kt < KT; ++kt) {
    const int cur = kt & 1;
    if (kt + 1 < KT) {
      stage(cur ^ 1, kt + 1);
      wait_async_le4();  // <=4 outstanding == next tile only => cur landed
    } else {
      wait_async_0();
    }
    __syncthreads();
    compute(cur);
    __syncthreads();  // everyone done reading buf before it is re-staged
  }

#pragma unroll
  for (int mi = 0; mi < 2; ++mi)
#pragma unroll
    for (int ni = 0; ni < 4; ++ni)
#pragma unroll
      for (int r = 0; r < 8; ++r) {
        const int row = mBlk + m0w + 16 * mi + r + 8 * hi;
        const int col = nBlk + n0w + 16 * ni + l16;
        C[(size_t)row * N + col] = (OutT)acc[mi][ni][r];
      }
}

// ---------------------------------------------------------------------------
// Kernel 2: per (head, chunk) partial state  T = phi(K)^T @ V  (64x64 f32).
// qkvh layout: [2048][3072] f16; k at col 1024+h*64, v at 2048+h*64.
// ---------------------------------------------------------------------------
__global__ __launch_bounds__(128) void chunk_kv(const _Float16* __restrict__ qkvh,
                                                float* __restrict__ ksum) {
  const int h = blockIdx.x;
  const int c = blockIdx.y;
  const int tid  = threadIdx.x;
  const int lane = tid & 31;
  const int wave = tid >> 5;
  const int l16  = lane & 15;
  const int hi   = (lane >> 4) & 1;

  __shared__ _Float16 Kt[64][72];  // [d][token]
  __shared__ _Float16 Vt[64][72];  // [d][token]

  const _Float16* base = qkvh + (size_t)(c * 64) * 3072 + h * 64;
#pragma unroll
  for (int i = tid; i < 512; i += 128) {  // 64 tokens x 8 groups of 8 halfs
    const int t = i >> 3;
    const int g = (i & 7) * 8;
    const v8h k8 = *(const v8h*)(base + (size_t)t * 3072 + 1024 + g);
    const v8h v8 = *(const v8h*)(base + (size_t)t * 3072 + 2048 + g);
#pragma unroll
    for (int j = 0; j < 8; ++j) {
      Kt[g + j][t] = (_Float16)(fmaxf((float)k8[j], 0.f) + 1e-6f);
      Vt[g + j][t] = v8[j];
    }
  }
  __syncthreads();

  const int m0w = (wave >> 1) * 32;  // dk
  const int n0w = (wave & 1) * 32;   // dv
  v8f acc[2][2] = {};
#pragma unroll
  for (int t0 = 0; t0 < 64; t0 += 32) {
    const v16h a0 = frag_A(&Kt[0][0], 72, m0w, t0, lane);
    const v16h a1 = frag_A(&Kt[0][0], 72, m0w + 16, t0, lane);
    const v16h b0 = frag_B(&Vt[0][0], 72, n0w, t0, lane);
    const v16h b1 = frag_B(&Vt[0][0], 72, n0w + 16, t0, lane);
    acc[0][0] = wmma_f16(a0, b0, acc[0][0]);
    acc[0][1] = wmma_f16(a0, b1, acc[0][1]);
    acc[1][0] = wmma_f16(a1, b0, acc[1][0]);
    acc[1][1] = wmma_f16(a1, b1, acc[1][1]);
  }

  float* dst = ksum + (size_t)(h * 32 + c) * 4096;  // [dk][dv]
#pragma unroll
  for (int mi = 0; mi < 2; ++mi)
#pragma unroll
    for (int ni = 0; ni < 2; ++ni)
#pragma unroll
      for (int r = 0; r < 8; ++r) {
        const int dk = m0w + 16 * mi + r + 8 * hi;
        const int dv = n0w + 16 * ni + l16;
        dst[(size_t)dk * 64 + dv] = acc[mi][ni][r];
      }
}

// ---------------------------------------------------------------------------
// Kernel 3: exclusive prefix over chunks; stored transposed [dv][dk] so the
// state rows are B-fragment rows (N=dv, contiguous K=dk) for kernel 4.
// ---------------------------------------------------------------------------
__global__ __launch_bounds__(256) void chunk_scan(const float* __restrict__ ksum,
                                                  float* __restrict__ sprev) {
  const int h = blockIdx.x;
  for (int e = threadIdx.x; e < 4096; e += 256) {
    const int dv = e & 63;
    const int dk = e >> 6;
    float run = 0.f;
    for (int c = 0; c < 32; ++c) {
      const size_t o = (size_t)(h * 32 + c) * 4096;
      sprev[o + (size_t)dv * 64 + dk] = run;  // exclusive, transposed
      run += ksum[o + (size_t)dk * 64 + dv];
    }
  }
}

// ---------------------------------------------------------------------------
// Kernel 4: out = mask(phiQ @ phiK^T) @ V + phiQ @ S_prev, normalized; f16 out.
// ---------------------------------------------------------------------------
__global__ __launch_bounds__(128) void chunk_out(
    const _Float16* __restrict__ qkvh, const float* __restrict__ sprev,
    const float* __restrict__ alpha_log, const float* __restrict__ beta_log,
    _Float16* __restrict__ attnh) {
  const int h = blockIdx.x;
  const int c = blockIdx.y;
  const int tid  = threadIdx.x;
  const int lane = tid & 31;
  const int wave = tid >> 5;
  const int l16  = lane & 15;
  const int hi   = (lane >> 4) & 1;

  __shared__ _Float16 Qs[64][72];  // [token][d] phi(q)
  __shared__ _Float16 Ks[64][72];  // [token][d] phi(k)
  __shared__ _Float16 Vt[64][72];  // [d][token] v transposed
  __shared__ _Float16 Ps[64][72];  // [tq][tk] masked scores

  const _Float16* base = qkvh + (size_t)(c * 64) * 3072 + h * 64;
#pragma unroll
  for (int i = tid; i < 512; i += 128) {
    const int t = i >> 3;
    const int g = (i & 7) * 8;
    const v8h q8 = *(const v8h*)(base + (size_t)t * 3072 + g);
    const v8h k8 = *(const v8h*)(base + (size_t)t * 3072 + 1024 + g);
    const v8h v8 = *(const v8h*)(base + (size_t)t * 3072 + 2048 + g);
#pragma unroll
    for (int j = 0; j < 8; ++j) {
      Qs[t][g + j] = (_Float16)(fmaxf((float)q8[j], 0.f) + 1e-6f);
      Ks[t][g + j] = (_Float16)(fmaxf((float)k8[j], 0.f) + 1e-6f);
      Vt[g + j][t] = v8[j];
    }
  }
  __syncthreads();

  const int m0w = (wave >> 1) * 32;  // tq / token rows
  const int n0w = (wave & 1) * 32;   // tk / dv cols

  // Phase 1: P = phiQ @ phiK^T with causal mask applied in C/D layout.
  {
    v8f p[2][2] = {};
#pragma unroll
    for (int d0 = 0; d0 < 64; d0 += 32) {
      const v16h a0 = frag_A(&Qs[0][0], 72, m0w, d0, lane);
      const v16h a1 = frag_A(&Qs[0][0], 72, m0w + 16, d0, lane);
      const v16h b0 = frag_B(&Ks[0][0], 72, n0w, d0, lane);
      const v16h b1 = frag_B(&Ks[0][0], 72, n0w + 16, d0, lane);
      p[0][0] = wmma_f16(a0, b0, p[0][0]);
      p[0][1] = wmma_f16(a0, b1, p[0][1]);
      p[1][0] = wmma_f16(a1, b0, p[1][0]);
      p[1][1] = wmma_f16(a1, b1, p[1][1]);
    }
#pragma unroll
    for (int mi = 0; mi < 2; ++mi)
#pragma unroll
      for (int ni = 0; ni < 2; ++ni)
#pragma unroll
        for (int r = 0; r < 8; ++r) {
          const int tq = m0w + 16 * mi + r + 8 * hi;
          const int tk = n0w + 16 * ni + l16;
          Ps[tq][tk] = (_Float16)((tk <= tq) ? p[mi][ni][r] : 0.f);
        }
  }
  __syncthreads();

  // Phase 2: out = P @ V + phiQ @ S_prev^T
  v8f o[2][2] = {};
#pragma unroll
  for (int t0 = 0; t0 < 64; t0 += 32) {
    const v16h a0 = frag_A(&Ps[0][0], 72, m0w, t0, lane);
    const v16h a1 = frag_A(&Ps[0][0], 72, m0w + 16, t0, lane);
    const v16h b0 = frag_B(&Vt[0][0], 72, n0w, t0, lane);
    const v16h b1 = frag_B(&Vt[0][0], 72, n0w + 16, t0, lane);
    o[0][0] = wmma_f16(a0, b0, o[0][0]);
    o[0][1] = wmma_f16(a0, b1, o[0][1]);
    o[1][0] = wmma_f16(a1, b0, o[1][0]);
    o[1][1] = wmma_f16(a1, b1, o[1][1]);
  }
  const float* S = sprev + (size_t)(h * 32 + c) * 4096;  // [dv][dk] f32
#pragma unroll
  for (int d0 = 0; d0 < 64; d0 += 32) {
    const v16h a0 = frag_A(&Qs[0][0], 72, m0w, d0, lane);
    const v16h a1 = frag_A(&Qs[0][0], 72, m0w + 16, d0, lane);
    const v16h b0 = frag_B_f32(S, 64, n0w, d0, lane);
    const v16h b1 = frag_B_f32(S, 64, n0w + 16, d0, lane);
    o[0][0] = wmma_f16(a0, b0, o[0][0]);
    o[0][1] = wmma_f16(a0, b1, o[0][1]);
    o[1][0] = wmma_f16(a1, b0, o[1][0]);
    o[1][1] = wmma_f16(a1, b1, o[1][1]);
  }

  // SteLLA normalization: denom = max(sp(a)*sp(b)*sqrt(64)*pos, eps)
  const float sa    = __logf(1.f + __expf(alpha_log[h])) + 1e-6f;
  const float sb    = __logf(1.f + __expf(beta_log[h])) + 1e-6f;
  const float scale = sa * sb * 8.0f;
#pragma unroll
  for (int mi = 0; mi < 2; ++mi)
#pragma unroll
    for (int ni = 0; ni < 2; ++ni)
#pragma unroll
      for (int r = 0; r < 8; ++r) {
        const int t      = m0w + 16 * mi + r + 8 * hi;
        const int dv     = n0w + 16 * ni + l16;
        const int n_glob = c * 64 + t;
        const float denom = fmaxf(scale * (float)(n_glob + 1), 1e-6f);
        attnh[(size_t)n_glob * 1024 + h * 64 + dv] =
            (_Float16)(o[mi][ni][r] / denom);
      }
}

// ---------------------------------------------------------------------------
// Host-side launcher
// ---------------------------------------------------------------------------
extern "C" void kernel_launch(void* const* d_in, const int* in_sizes, int n_in,
                              void* d_out, int out_size, void* d_ws,
                              size_t ws_size, hipStream_t stream) {
  (void)in_sizes; (void)n_in; (void)out_size; (void)ws_size;

  const float* x         = (const float*)d_in[0];  // [1,2048,1024]
  const float* qkv_w     = (const float*)d_in[1];  // [3072,1024]
  const float* proj_w    = (const float*)d_in[2];  // [1024,1024]
  const float* alpha_log = (const float*)d_in[3];  // [16]
  const float* beta_log  = (const float*)d_in[4];  // [16]
  float*       out       = (float*)d_out;          // [1,2048,1024]

  // Workspace carve-up (all 16B aligned):
  // xh(4MB) wqkvh(6MB) wprojh(2MB) qkvh(12MB) ksum(8MB) sprev(8MB) attnh(4MB)
  _Float16* xh     = (_Float16*)d_ws;                 // 2048*1024
  _Float16* wqkvh  = xh + (size_t)2048 * 1024;        // 3072*1024
  _Float16* wprojh = wqkvh + (size_t)3072 * 1024;     // 1024*1024
  _Float16* qkvh   = wprojh + (size_t)1024 * 1024;    // 2048*3072
  float*    ksum   = (float*)(qkvh + (size_t)2048 * 3072);  // 16*32*64*64
  float*    sprev  = ksum + (size_t)16 * 32 * 64 * 64;      // 16*32*64*64
  _Float16* attnh  = (_Float16*)(sprev + (size_t)16 * 32 * 64 * 64);  // 2048*1024

  const dim3 b128(128);
  const dim3 b256(256);

  // 0) one-shot f32 -> f16 conversion of inputs
  f32_to_f16<<<dim3((2048 * 1024 / 4 + 255) / 256), b256, 0, stream>>>(
      x, xh, 2048 * 1024 / 4);
  f32_to_f16<<<dim3((3072 * 1024 / 4 + 255) / 256), b256, 0, stream>>>(
      qkv_w, wqkvh, 3072 * 1024 / 4);
  f32_to_f16<<<dim3((1024 * 1024 / 4 + 255) / 256), b256, 0, stream>>>(
      proj_w, wprojh, 1024 * 1024 / 4);

  // 1) qkv = x @ qkv_w^T  (f16 out)
  gemm_xwt_f16<_Float16><<<dim3(3072 / 128, 2048 / 128), b256, 0, stream>>>(
      xh, wqkvh, qkvh, 2048, 3072, 1024);
  // 2) per-chunk KV states
  chunk_kv<<<dim3(16, 32), b128, 0, stream>>>(qkvh, ksum);
  // 3) exclusive prefix over chunks
  chunk_scan<<<dim3(16), b256, 0, stream>>>(ksum, sprev);
  // 4) intra + inter chunk attention output (normalized, f16 out)
  chunk_out<<<dim3(16, 32), b128, 0, stream>>>(qkvh, sprev, alpha_log,
                                               beta_log, attnh);
  // 5) out = attn @ proj_w^T  (f32 out)
  gemm_xwt_f16<float><<<dim3(1024 / 128, 2048 / 128), b256, 0, stream>>>(
      attnh, wprojh, out, 2048, 1024, 1024);
}
